// CAM_Module_6519760355966
// MI455X (gfx1250) — matrix operand order repaired and checked by reading, not compiled, not run on hardware
//
#include <hip/hip_runtime.h>

// ---------------------------------------------------------------------------
// CAM_Module channel attention for MI455X (gfx1250, wave32, WMMA).
//   x: (16, 512, 64, 64) f32,  gamma: (1,) f32
//   energy = X X^T  (bf16 WMMA, f32 accum)
//   attn   = softmax(rowmax - energy)  == exp(rowmin - e)/sum
//   out    = gamma * (attn X) + x
// Scratch layout in d_ws (needs ~92.3 MB):
//   [0, 67MB)      xb   : bf16 copy of x       (16*512*4096 ushort)
//   [+,  +16.8MB)  ener : energy f32           (16*512*512 float)
//   [+,  +8.4MB)   attn : attention bf16       (16*512*512 ushort)
// Data movement: double-buffered LDS tiles filled with
// GLOBAL_LOAD_ASYNC_TO_LDS_B128 (ASYNCcnt), bypassing VGPR staging.
// ---------------------------------------------------------------------------

#define B_  16
#define C_  512
#define N_  4096   // 64*64 spatial

typedef __attribute__((ext_vector_type(16))) __bf16       v16bf;
typedef __attribute__((ext_vector_type(8)))  float        v8f;
typedef __attribute__((ext_vector_type(4)))  unsigned int u32x4;
typedef __attribute__((ext_vector_type(2)))  unsigned int u32x2;
typedef __attribute__((ext_vector_type(4)))  float        f32x4;
typedef __attribute__((ext_vector_type(4)))  int          i32x4;

union FragBF {            // one WMMA bf16 A/B operand (16 halves / lane)
    v16bf          v;
    unsigned short s[16];
    u32x4          q[2];
};

#ifndef __has_builtin
#define __has_builtin(x) 0
#endif

#if __has_builtin(__builtin_amdgcn_global_load_async_to_lds_b128)
#define ASYNC_LDS 1
#else
#define ASYNC_LDS 0
#endif

typedef __attribute__((address_space(1))) i32x4 gbl_i32x4;  // global int4*
typedef __attribute__((address_space(3))) i32x4 lds_i32x4;  // LDS int4*

// 16B global -> LDS copy (async path bypasses VGPRs, tracked by ASYNCcnt)
__device__ __forceinline__ void cp16(void* lds_dst, const void* gsrc) {
#if ASYNC_LDS
    __builtin_amdgcn_global_load_async_to_lds_b128(
        (gbl_i32x4*)gsrc, (lds_i32x4*)lds_dst, 0, 0);
#else
    *reinterpret_cast<u32x4*>(lds_dst) = *reinterpret_cast<const u32x4*>(gsrc);
#endif
}

#if ASYNC_LDS
#  if __has_builtin(__builtin_amdgcn_s_wait_asynccnt)
#    define WAIT_ASYNC(n) __builtin_amdgcn_s_wait_asynccnt(n)
#  else
#    define WAIT_ASYNC(n) asm volatile("s_wait_asynccnt " #n ::: "memory")
#  endif
#else
#  define WAIT_ASYNC(n) ((void)0)
#endif

__device__ __forceinline__ unsigned short f32_to_bf16(float f) {
    unsigned int u = __float_as_uint(f);
    u += 0x7FFFu + ((u >> 16) & 1u);     // round-to-nearest-even
    return (unsigned short)(u >> 16);
}

// ---------------------------------------------------------------------------
// 1) fp32 -> bf16 conversion, 16B vector loads
// ---------------------------------------------------------------------------
__global__ __launch_bounds__(256)
void k_cvt(const float* __restrict__ x, unsigned short* __restrict__ xb, int n4) {
    int i      = blockIdx.x * blockDim.x + threadIdx.x;
    int stride = gridDim.x * blockDim.x;
    for (; i < n4; i += stride) {
        f32x4 v = reinterpret_cast<const f32x4*>(x)[i];
        u32x2 o;
        o.x = (unsigned int)f32_to_bf16(v.x) | ((unsigned int)f32_to_bf16(v.y) << 16);
        o.y = (unsigned int)f32_to_bf16(v.z) | ((unsigned int)f32_to_bf16(v.w) << 16);
        reinterpret_cast<u32x2*>(xb)[i] = o;
    }
}

// ---------------------------------------------------------------------------
// 2) energy[b] = X X^T ; 64x64 tile / block, 8 waves, 2 subtiles per wave,
//    double-buffered async LDS pipeline
// ---------------------------------------------------------------------------
#define LDA 40   // padded LDS row stride (halves); 80B = multiple of 16B

__global__ __launch_bounds__(256)
void k_energy(const unsigned short* __restrict__ xb, float* __restrict__ energy) {
    __shared__ __align__(16) unsigned short lA[2][64][LDA];
    __shared__ __align__(16) unsigned short lB[2][64][LDA];

    const int b  = blockIdx.z;
    const int mb = blockIdx.y * 64;
    const int nb = blockIdx.x * 64;
    const unsigned short* Xb = xb + (size_t)b * C_ * N_;

    const int t    = threadIdx.x;
    const int lane = t & 31;
    const int wave = t >> 5;         // 0..7
    const int sr   = wave >> 1;      // subtile row 0..3
    const int sc0  = (wave & 1) * 2; // subtile cols {sc0, sc0+1}

    const int lr = t >> 2;           // load row 0..63
    const int lc = (t & 3) * 8;      // load col {0,8,16,24}
    const unsigned short* gA = Xb + (size_t)(mb + lr) * N_ + lc;
    const unsigned short* gB = Xb + (size_t)(nb + lr) * N_ + lc;

    const int arow = sr * 16 + (lane & 15);
    const int kh   = lane >> 4;          // A K-half select
    const int bko  = (lane >> 4) * 16;   // B K-offset (contiguous 16)
    const int br0  = sc0 * 16 + (lane & 15);
    const int br1  = br0 + 16;

    auto issue = [&](int k) {            // 2 async copies per thread per stage
        const int s = k & 1;
        cp16(&lA[s][lr][lc], gA + k * 32);
        cp16(&lB[s][lr][lc], gB + k * 32);
    };

    constexpr int K = N_ / 32;           // 128 K-steps
    issue(0);
    issue(1);

    v8f acc0 = {}; v8f acc1 = {};
    for (int k = 0; k < K; ++k) {
        if (k + 1 < K) { WAIT_ASYNC(2); } else { WAIT_ASYNC(0); }
        __syncthreads();                 // stage-k tiles resident for all waves

        const int s = k & 1;
        FragBF a, b0, b1;
        // A 16x32 bf16: lane = row, K = kh*16+0..15 contiguous
        a.q[0] = *reinterpret_cast<const u32x4*>(&lA[s][arow][kh * 16]);
        a.q[1] = *reinterpret_cast<const u32x4*>(&lA[s][arow][kh * 16 + 8]);
        // B 32x16 bf16: lane = col (= X row nb+n), K = bko+0..15 contiguous
        b0.q[0] = *reinterpret_cast<const u32x4*>(&lB[s][br0][bko]);
        b0.q[1] = *reinterpret_cast<const u32x4*>(&lB[s][br0][bko + 8]);
        b1.q[0] = *reinterpret_cast<const u32x4*>(&lB[s][br1][bko]);
        b1.q[1] = *reinterpret_cast<const u32x4*>(&lB[s][br1][bko + 8]);

        acc0 = __builtin_amdgcn_wmma_f32_16x16x32_bf16(false, a.v, false, b0.v,
                                                       (short)0, acc0, false, false);
        acc1 = __builtin_amdgcn_wmma_f32_16x16x32_bf16(false, a.v, false, b1.v,
                                                       (short)0, acc1, false, false);
        __syncthreads();                 // all waves done reading buffer k&1
        if (k + 2 < K) issue(k + 2);     // refill the buffer just freed
    }

    // C/D layout: VGPR i -> M = i + 8*(lane/16), N = lane%16
    float* E = energy + (size_t)b * C_ * C_;
    const int row = mb + sr * 16 + 8 * (lane >> 4);
    const int c0  = nb + sc0 * 16 + (lane & 15);
#pragma unroll
    for (int i = 0; i < 8; ++i) {
        E[(size_t)(row + i) * C_ + c0]      = acc0[i];
        E[(size_t)(row + i) * C_ + c0 + 16] = acc1[i];
    }
}

// ---------------------------------------------------------------------------
// 3) attn = softmax(rowmax - e) = exp(rowmin - e) / sum ; one block per row
// ---------------------------------------------------------------------------
__global__ __launch_bounds__(256)
void k_softmax(const float* __restrict__ energy, unsigned short* __restrict__ attn) {
    __shared__ float red[256];
    const int    t = threadIdx.x;
    const size_t r = (size_t)blockIdx.x * C_;

    float v0 = energy[r + t];
    float v1 = energy[r + t + 256];

    red[t] = fminf(v0, v1);
    __syncthreads();
    for (int s = 128; s > 0; s >>= 1) {
        if (t < s) red[t] = fminf(red[t], red[t + s]);
        __syncthreads();
    }
    const float rowmin = red[0];
    __syncthreads();

    float p0 = __expf(rowmin - v0);
    float p1 = __expf(rowmin - v1);
    red[t] = p0 + p1;
    __syncthreads();
    for (int s = 128; s > 0; s >>= 1) {
        if (t < s) red[t] += red[t + s];
        __syncthreads();
    }
    const float inv = 1.0f / red[0];

    attn[r + t]       = f32_to_bf16(p0 * inv);
    attn[r + t + 256] = f32_to_bf16(p1 * inv);
}

// ---------------------------------------------------------------------------
// 4) out = gamma * (attn @ X) + x ; 64x64 tile / block, same async pipeline
// ---------------------------------------------------------------------------
#define LDB 72   // K-major B tile row stride (halves); 144B = multiple of 16B

__global__ __launch_bounds__(256)
void k_out(const unsigned short* __restrict__ attn,
           const unsigned short* __restrict__ xb,
           const float* __restrict__ x,
           const float* __restrict__ gamma,
           float* __restrict__ out) {
    __shared__ __align__(16) unsigned short lA[2][64][LDA];   // attn tile 64 x 32
    __shared__ __align__(16) unsigned short lB[2][32][LDB];   // X tile 32k x 64n

    const int b  = blockIdx.z;
    const int mb = blockIdx.y * 64;       // channel rows
    const int nb = blockIdx.x * 64;       // spatial cols
    const unsigned short* A  = attn + (size_t)b * C_ * C_;
    const unsigned short* Xb = xb   + (size_t)b * C_ * N_;

    const int t    = threadIdx.x;
    const int lane = t & 31;
    const int wave = t >> 5;
    const int sr   = wave >> 1;
    const int sc0  = (wave & 1) * 2;

    const int lrA = t >> 2;           // 0..63
    const int lcA = (t & 3) * 8;      // 0,8,16,24
    const int lrB = t >> 3;           // 0..31 (K)
    const int lcB = (t & 7) * 8;      // 0..56 (N)
    const unsigned short* gA = A  + (size_t)(mb + lrA) * C_ + lcA;
    const unsigned short* gB = Xb + (size_t)lrB * N_ + nb + lcB;

    const int arow = sr * 16 + (lane & 15);
    const int kh   = lane >> 4;
    const int bk   = (lane >> 4) * 16;
    const int n0   = sc0 * 16 + (lane & 15);

    auto issue = [&](int k) {
        const int s = k & 1;
        cp16(&lA[s][lrA][lcA], gA + k * 32);
        cp16(&lB[s][lrB][lcB], gB + (size_t)k * 32 * N_);
    };

    constexpr int K = C_ / 32;            // 16 K-steps
    issue(0);
    issue(1);

    v8f acc0 = {}; v8f acc1 = {};
    for (int k = 0; k < K; ++k) {
        if (k + 1 < K) { WAIT_ASYNC(2); } else { WAIT_ASYNC(0); }
        __syncthreads();

        const int s = k & 1;
        FragBF a, b0, b1;
        a.q[0] = *reinterpret_cast<const u32x4*>(&lA[s][arow][kh * 16]);
        a.q[1] = *reinterpret_cast<const u32x4*>(&lA[s][arow][kh * 16 + 8]);
#pragma unroll
        for (int i = 0; i < 16; ++i) {    // column gather from K-major tile
            b0.s[i] = lB[s][bk + i][n0];
            b1.s[i] = lB[s][bk + i][n0 + 16];
        }

        acc0 = __builtin_amdgcn_wmma_f32_16x16x32_bf16(false, a.v, false, b0.v,
                                                       (short)0, acc0, false, false);
        acc1 = __builtin_amdgcn_wmma_f32_16x16x32_bf16(false, a.v, false, b1.v,
                                                       (short)0, acc1, false, false);
        __syncthreads();
        if (k + 2 < K) issue(k + 2);
    }

    const float g   = gamma[0];
    const int   row = mb + sr * 16 + 8 * (lane >> 4);
    const int   c0  = nb + sc0 * 16 + (lane & 15);
    const size_t base = (size_t)b * C_ * N_;
#pragma unroll
    for (int i = 0; i < 8; ++i) {
        size_t i0 = base + (size_t)(row + i) * N_ + c0;
        out[i0]      = g * acc0[i] + x[i0];
        out[i0 + 16] = g * acc1[i] + x[i0 + 16];
    }
}

// ---------------------------------------------------------------------------
extern "C" void kernel_launch(void* const* d_in, const int* in_sizes, int n_in,
                              void* d_out, int out_size, void* d_ws, size_t ws_size,
                              hipStream_t stream) {
    (void)in_sizes; (void)n_in; (void)out_size; (void)ws_size;

    const float* x     = (const float*)d_in[0];
    const float* gamma = (const float*)d_in[1];
    float*       out   = (float*)d_out;

    const size_t XB_BYTES = (size_t)B_ * C_ * N_ * sizeof(unsigned short); // 67 MB
    const size_t E_BYTES  = (size_t)B_ * C_ * C_ * sizeof(float);          // 16.8 MB

    unsigned short* xb   = (unsigned short*)d_ws;
    float*          ener = (float*)((char*)d_ws + XB_BYTES);
    unsigned short* attn = (unsigned short*)((char*)d_ws + XB_BYTES + E_BYTES);

    const int n4 = (B_ * C_ * N_) / 4;                 // 8,388,608 float4s
    k_cvt<<<dim3(8192), dim3(256), 0, stream>>>(x, xb, n4);

    k_energy<<<dim3(C_ / 64, C_ / 64, B_), dim3(256), 0, stream>>>(xb, ener);

    k_softmax<<<dim3(B_ * C_), dim3(256), 0, stream>>>(ener, attn);

    k_out<<<dim3(N_ / 64, C_ / 64, B_), dim3(256), 0, stream>>>(attn, xb, x, gamma, out);
}
